// StackLSTM_53429393162938
// MI455X (gfx1250) — matrix operand order repaired and checked by reading, not compiled, not run on hardware
//
#include <hip/hip_runtime.h>
#include <hip/hip_bf16.h>
#include <math.h>

// ---------------------------------------------------------------------------
// Stack-LSTM on MI455X (gfx1250, wave32, WMMA 16x16x32 f16 -> f32)
//
// Sizes: T=128, B=64, VOCAB=8000, E=128, H=256, D=128
// ---------------------------------------------------------------------------

#define T_SEQ 128
#define BATCH 64
#define VOCAB 8000
#define EDIM  128
#define HDIM  256
#define DDIM  128
#define KCAT  512      // E + D + H = 128+128+256
#define NGATE 1024     // 4*H
#define NSML  160      // D + pop + push padded to 16-multiple

typedef __attribute__((ext_vector_type(8)))  _Float16 v8h;
typedef __attribute__((ext_vector_type(16))) _Float16 v16h;
typedef __attribute__((ext_vector_type(8)))  float    v8f;
typedef __attribute__((ext_vector_type(4)))  float    v4f;

__device__ __forceinline__ v16h cat16(v8h lo, v8h hi) {
  return __builtin_shufflevector(lo, hi, 0,1,2,3,4,5,6,7,8,9,10,11,12,13,14,15);
}

// A-fragment (16x32 f16, M x K) per ISA 7.12.2:
// lanes 0-15: row M=lane,   K = k0+0..7 (v0..3) and k0+16..23 (v4..7)
// lanes 16-31: row M=lane-16, K = k0+8..15 and k0+24..31
__device__ __forceinline__ v16h loadA(const _Float16* base, int stride,
                                      int m0, int k0, int lane) {
  int row = m0 + (lane & 15);
  int kb  = k0 + ((lane >> 4) << 3);
  const _Float16* p = base + row * stride + kb;
  return cat16(*(const v8h*)p, *(const v8h*)(p + 16));
}

// B-fragment (32x16 f16, K x N) from row-major W[n][k] (i.e. B^T in memory):
// lane holds column N = n0 + (lane&15); lanes 0-15 hold K=k0+0..7 and k0+16..23,
// lanes 16-31 hold K=k0+8..15 and k0+24..31 (2 halfs per VGPR).
__device__ __forceinline__ v16h loadB(const _Float16* W, int strideK,
                                      int n0, int k0, int lane) {
  int row = n0 + (lane & 15);
  int kb  = k0 + ((lane >> 4) << 3);
  const _Float16* p = W + (size_t)row * strideK + kb;
  return cat16(*(const v8h*)p, *(const v8h*)(p + 16));
}

__device__ __forceinline__ v8f wmma_f16(v16h a, v16h b, v8f c) {
  return __builtin_amdgcn_wmma_f32_16x16x32_f16(false, a, false, b,
                                                (short)0, c, false, false);
}

__device__ __forceinline__ float sigm(float x) { return 1.f / (1.f + __expf(-x)); }

// ---------------------------------------------------------------------------
// Prep kernels: f32 -> f16 casts / gathers / weight fusion
// ---------------------------------------------------------------------------
__global__ void k_embed(const int* __restrict__ x, const float* __restrict__ emb,
                        _Float16* __restrict__ exf) {
  int i = blockIdx.x * 256 + threadIdx.x;           // over T*B*E
  if (i >= T_SEQ * BATCH * EDIM) return;
  int tb = i >> 7, e = i & 127;
  exf[i] = (_Float16)emb[(size_t)x[tb] * EDIM + e];
}

__global__ void k_wcat(const float* __restrict__ W_ih, const float* __restrict__ W_hh,
                       _Float16* __restrict__ Wcat) {
  int i = blockIdx.x * 256 + threadIdx.x;           // over 1024*512
  if (i >= NGATE * KCAT) return;
  int n = i >> 9, k = i & 511;
  float v = (k < 256) ? W_ih[n * 256 + k] : W_hh[n * 256 + (k - 256)];
  Wcat[i] = (_Float16)v;
}

__global__ void k_wsmall(const float* __restrict__ W_val, const float* __restrict__ W_pop,
                         const float* __restrict__ W_push, _Float16* __restrict__ Ws) {
  int i = blockIdx.x * 256 + threadIdx.x;           // over 160*256
  if (i >= NSML * HDIM) return;
  int n = i >> 8, k = i & 255;
  float v = (n < 128) ? W_val[n * 256 + k]
          : (n == 128) ? W_pop[k]
          : (n == 129) ? W_push[k] : 0.f;
  Ws[i] = (_Float16)v;
}

__global__ void k_cast(const float* __restrict__ src, _Float16* __restrict__ dst, int n) {
  int i = blockIdx.x * 256 + threadIdx.x;
  if (i < n) dst[i] = (_Float16)src[i];
}

// ---------------------------------------------------------------------------
// Persistent recurrent scan kernel: 1 block, 512 threads (16 waves), one WGP.
// All recurrent state lives in LDS. hs (f16, [b][t][h]) and Vbuf (f32,
// [t][b][d]) go to workspace global memory.
// ---------------------------------------------------------------------------
#define SA 520    // act row stride (halfs): [64][512] padded (+8)
#define SG 1032   // gates row stride (halfs): [64][1024] padded (+8)
#define SS 129    // s / w row stride (floats): [64][128] padded (+1)

#define OFF_ACT   0
#define OFF_CX    66560                   // 64*520*2
#define OFF_S     (OFF_CX + 65536)        // 64*256*4
#define OFF_GATES (OFF_S + 33024)         // 64*129*4
#define OFF_W     OFF_GATES               // aliases gates (dead by then)
#define OFF_POP   (OFF_GATES + 132096)    // 64*1032*2
#define OFF_PUSH  (OFF_POP + 256)
#define SMEM_BYTES (OFF_PUSH + 256)       // 297,728 B  (< 320 KB)

__global__ __launch_bounds__(512)
void scan_kernel(const _Float16* __restrict__ exf,
                 const _Float16* __restrict__ Wcat,
                 const _Float16* __restrict__ Wsmall,
                 const float* __restrict__ b_ih, const float* __restrict__ b_hh,
                 const float* __restrict__ b_val, const float* __restrict__ b_pop,
                 const float* __restrict__ b_push,
                 _Float16* __restrict__ hsA, float* __restrict__ Vbuf) {
  extern __shared__ __align__(16) char smem[];
  _Float16* act   = (_Float16*)(smem + OFF_ACT);   // [64][SA]: 0..127 ex, 128..255 rx, 256..511 hx
  float*    cxA   = (float*)(smem + OFF_CX);       // [64][256]
  float*    sArr  = (float*)(smem + OFF_S);        // [64][SS]
  _Float16* gates = (_Float16*)(smem + OFF_GATES); // [64][SG]
  float*    wArr  = (float*)(smem + OFF_W);        // [64][SS], aliases gates
  float*    popA  = (float*)(smem + OFF_POP);
  float*    pushA = (float*)(smem + OFF_PUSH);

  const int tid  = threadIdx.x;
  const int wid  = tid >> 5;
  const int lane = tid & 31;

  // ---- init persistent state ----
  for (int i = tid; i < BATCH * SA; i += 512) act[i] = (_Float16)0.f;
  for (int i = tid; i < BATCH * HDIM; i += 512) cxA[i] = 0.f;
  for (int i = tid; i < BATCH * SS; i += 512) sArr[i] = 0.f;
  __syncthreads();
  for (int i = tid; i < BATCH * DDIM; i += 512) {
    int b = i >> 7, d = i & 127;
    act[b * SA + 128 + d] = (_Float16)1.0f;        // rx init = ones
  }
  __syncthreads();

  for (int t = 0; t < T_SEQ; ++t) {
    // 1) stage embedded token into act[:, 0:128]
    for (int i = tid; i < BATCH * EDIM; i += 512) {
      int b = i >> 7, e = i & 127;
      act[b * SA + e] = exf[(size_t)t * (BATCH * EDIM) + i];
    }
    __syncthreads();

    // 2) gate GEMM: [64 x 512] @ Wcat^T -> gates [64 x 1024] (f16, +bias)
    for (int id = wid; id < 4 * 64; id += 16) {
      int m0 = (id & 3) * 16, n0 = (id >> 2) * 16;
      v8f acc = {0.f,0.f,0.f,0.f,0.f,0.f,0.f,0.f};
      #pragma unroll
      for (int k0 = 0; k0 < KCAT; k0 += 32) {
        v16h a = loadA(act, SA, m0, k0, lane);
        v16h b = loadB(Wcat, KCAT, n0, k0, lane);
        acc = wmma_f16(a, b, acc);
      }
      int n = n0 + (lane & 15);
      float bias = b_ih[n] + b_hh[n];
      int mhi = (lane >> 4) << 3;
      #pragma unroll
      for (int r = 0; r < 8; ++r)
        gates[(m0 + r + mhi) * SG + n] = (_Float16)(acc[r] + bias);
    }
    __syncthreads();

    // 3) LSTM elementwise; write new hx into act[:,256:512] and hs_f16
    for (int i = tid; i < BATCH * HDIM; i += 512) {
      int b = i >> 8, h = i & 255;
      const _Float16* g = gates + b * SG;
      float ig = (float)g[h],        fg = (float)g[256 + h];
      float gg = (float)g[512 + h],  og = (float)g[768 + h];
      float c  = sigm(fg) * cxA[i] + sigm(ig) * tanhf(gg);
      float hv = sigm(og) * tanhf(c);
      cxA[i] = c;
      act[b * SA + 256 + h] = (_Float16)hv;
      hsA[((size_t)b * T_SEQ + t) * HDIM + h] = (_Float16)hv;
    }
    __syncthreads();

    // 4) head GEMM: hx_new[64x256] @ [W_val;W_pop;W_push]^T -> vals/pop/push
    for (int id = wid; id < 4 * 10; id += 16) {
      int m0 = (id & 3) * 16, n0 = (id >> 2) * 16;
      v8f acc = {0.f,0.f,0.f,0.f,0.f,0.f,0.f,0.f};
      #pragma unroll
      for (int k0 = 0; k0 < HDIM; k0 += 32) {
        v16h a = loadA(act + 256, SA, m0, k0, lane);
        v16h b = loadB(Wsmall, HDIM, n0, k0, lane);
        acc = wmma_f16(a, b, acc);
      }
      int n = n0 + (lane & 15);
      int mhi = (lane >> 4) << 3;
      if (n < DDIM) {
        float bias = b_val[n];
        #pragma unroll
        for (int r = 0; r < 8; ++r) {
          int bb = m0 + r + mhi;
          Vbuf[(size_t)t * (BATCH * DDIM) + bb * DDIM + n] = sigm(acc[r] + bias);
        }
      } else if (n == DDIM) {
        #pragma unroll
        for (int r = 0; r < 8; ++r) popA[m0 + r + mhi] = sigm(acc[r] + b_pop[0]);
      } else if (n == DDIM + 1) {
        #pragma unroll
        for (int r = 0; r < 8; ++r) pushA[m0 + r + mhi] = sigm(acc[r] + b_push[0]);
      }
    }
    __syncthreads();

    // 5) differentiable stack update (thread-per-batch, serial top-down scans)
    if (tid < BATCH) {
      int b = tid;
      float pop = popA[b], push = pushA[b];
      float* sb = sArr + b * SS;
      float run = 0.f;
      for (int i = t - 1; i >= 0; --i) {       // suffix over OLD strengths
        float si = sb[i];
        float sn = fmaxf(si - fmaxf(pop - run, 0.f), 0.f);
        run += si;
        sb[i] = sn;
      }
      sb[t] = push;                             // push at slot t
      float* wb = wArr + b * SS;
      float run2 = 0.f;
      for (int i = t; i >= 0; --i) {            // read weights from NEW strengths
        float sn = sb[i];
        wb[i] = fminf(sn, fmaxf(1.f - run2, 0.f));
        run2 += sn;
      }
    }
    __syncthreads();

    // 6) stack read: rx[b,d] = sum_{t'<=t} w[b,t'] * Vbuf[t',b,d]
    //    vectorized: 4 x float4 loads + vector FMAs per time step
    {
      int b = tid >> 3, d0 = (tid & 7) << 4;    // 64 batches x 8 groups of 16 d
      v4f a0 = {0.f,0.f,0.f,0.f}, a1 = a0, a2 = a0, a3 = a0;
      for (int tp = 0; tp <= t; ++tp) {
        float wv = wArr[b * SS + tp];
        const v4f* vp = (const v4f*)(Vbuf + (size_t)tp * (BATCH * DDIM) + b * DDIM + d0);
        a0 += wv * vp[0];
        a1 += wv * vp[1];
        a2 += wv * vp[2];
        a3 += wv * vp[3];
      }
      _Float16* rdst = act + b * SA + 128 + d0;
      #pragma unroll
      for (int j = 0; j < 4; ++j) rdst[j]      = (_Float16)a0[j];
      #pragma unroll
      for (int j = 0; j < 4; ++j) rdst[4 + j]  = (_Float16)a1[j];
      #pragma unroll
      for (int j = 0; j < 4; ++j) rdst[8 + j]  = (_Float16)a2[j];
      #pragma unroll
      for (int j = 0; j < 4; ++j) rdst[12 + j] = (_Float16)a3[j];
    }
    __syncthreads();
  }
}

// ---------------------------------------------------------------------------
// Output GEMM: out[b,v,t] = sum_h hs[b,t,h] * W_out[v,h] + b_out[v]
// 64 batched GEMMs of [128 x 256] x [256 x 8000]. Block = 8 waves = 256 thr,
// computes one 16-row t-tile x 256 v-columns for one batch element.
// Epilogue uses explicit float4 stores: each lane writes 2 x b128, and the
// lane L / L+16 pair forms a contiguous 64B segment of the output row.
// ---------------------------------------------------------------------------
#define ALDS_STRIDE 264

__global__ __launch_bounds__(256)
void out_gemm(const _Float16* __restrict__ hsA, const _Float16* __restrict__ Wout,
              const float* __restrict__ b_out, float* __restrict__ out) {
  __shared__ __align__(16) _Float16 At[16 * ALDS_STRIDE];
  const int b  = blockIdx.z;
  const int mt = blockIdx.y;
  const int bx = blockIdx.x;
  const int tid = threadIdx.x, wid = tid >> 5, lane = tid & 31;

  // stage A tile [16 x 256] from hs_f16[b][mt*16 .. +15][:]
  const _Float16* src = hsA + ((size_t)b * T_SEQ + mt * 16) * HDIM;
  for (int i = tid; i < 16 * 32; i += 256) {      // 512 chunks of 8 halfs
    int row = i >> 5, c8 = (i & 31) << 3;
    *(v8h*)(At + row * ALDS_STRIDE + c8) = *(const v8h*)(src + row * HDIM + c8);
  }
  __syncthreads();

  #pragma unroll
  for (int j = 0; j < 2; ++j) {
    int nt = bx * 16 + wid * 2 + j;
    if (nt >= VOCAB / 16) continue;               // 500 real v-tiles
    int v0 = nt * 16;
    // warm L2 for the next tile's B rows
    if (j == 0 && (bx * 16 + wid * 2 + 1) < VOCAB / 16)
      __builtin_prefetch((const void*)(Wout + (size_t)(v0 + 16) * HDIM), 0, 1);
    v8f acc = {0.f,0.f,0.f,0.f,0.f,0.f,0.f,0.f};
    #pragma unroll
    for (int k0 = 0; k0 < HDIM; k0 += 32) {
      v16h a  = loadA(At, ALDS_STRIDE, 0, k0, lane);
      v16h bb = loadB(Wout, HDIM, v0, k0, lane);
      acc = wmma_f16(a, bb, acc);
    }
    int n = v0 + (lane & 15);
    float bias = b_out[n];
    int mhi = (lane >> 4) << 3;
    int t0 = mt * 16;
    v4f lo = {acc[0] + bias, acc[1] + bias, acc[2] + bias, acc[3] + bias};
    v4f hi = {acc[4] + bias, acc[5] + bias, acc[6] + bias, acc[7] + bias};
    float* dst = out + ((size_t)b * VOCAB + n) * T_SEQ + t0 + mhi;
    *(v4f*)dst       = lo;   // 16B-aligned: t0%16==0, mhi%8==0
    *(v4f*)(dst + 4) = hi;
  }
}

// ---------------------------------------------------------------------------
// Launcher. Workspace layout (bytes):
//   [0)          ex_f16   T*B*E halfs      = 2,097,152
//   [2,097,152)  Wcat_f16 1024*512 halfs   = 1,048,576
//   [3,145,728)  Wsml_f16 160*256 halfs    =    81,920
//   [3,227,648)  Wout_f16 8000*256 halfs   = 4,096,000
//   [7,323,648)  hs_f16   B*T*H halfs      = 4,194,304
//   [11,517,952) Vbuf f32 T*B*D floats     = 4,194,304
//   total: 15,712,256 B
// ---------------------------------------------------------------------------
extern "C" void kernel_launch(void* const* d_in, const int* in_sizes, int n_in,
                              void* d_out, int out_size, void* d_ws, size_t ws_size,
                              hipStream_t stream) {
  const int*   x      = (const int*)  d_in[0];
  const float* emb    = (const float*)d_in[1];
  const float* W_ih   = (const float*)d_in[2];
  const float* b_ih   = (const float*)d_in[3];
  const float* W_hh   = (const float*)d_in[4];
  const float* b_hh   = (const float*)d_in[5];
  const float* W_out  = (const float*)d_in[6];
  const float* b_out  = (const float*)d_in[7];
  const float* W_push = (const float*)d_in[8];
  const float* b_push = (const float*)d_in[9];
  const float* W_pop  = (const float*)d_in[10];
  const float* b_pop  = (const float*)d_in[11];
  const float* W_val  = (const float*)d_in[12];
  const float* b_val  = (const float*)d_in[13];
  float* out = (float*)d_out;

  char* ws = (char*)d_ws;
  _Float16* exf    = (_Float16*)(ws + 0);
  _Float16* Wcat   = (_Float16*)(ws + 2097152);
  _Float16* Wsmall = (_Float16*)(ws + 3145728);
  _Float16* Woutf  = (_Float16*)(ws + 3227648);
  _Float16* hsA    = (_Float16*)(ws + 7323648);
  float*    Vbuf   = (float*)   (ws + 11517952);

  // ---- prep ----
  k_embed<<<(T_SEQ * BATCH * EDIM + 255) / 256, 256, 0, stream>>>(x, emb, exf);
  k_wcat<<<(NGATE * KCAT + 255) / 256, 256, 0, stream>>>(W_ih, W_hh, Wcat);
  k_wsmall<<<(NSML * HDIM + 255) / 256, 256, 0, stream>>>(W_val, W_pop, W_push, Wsmall);
  k_cast<<<(VOCAB * HDIM + 255) / 256, 256, 0, stream>>>(W_out, Woutf, VOCAB * HDIM);

  // ---- sequential recurrence: one persistent workgroup ----
  scan_kernel<<<1, 512, SMEM_BYTES, stream>>>(exf, Wcat, Wsmall,
                                              b_ih, b_hh, b_val, b_pop, b_push,
                                              hsA, Vbuf);

  // ---- output GEMM across the whole device ----
  dim3 grid(32, T_SEQ / 16, BATCH);   // (v-blocks of 256, t-tiles, batch)
  out_gemm<<<grid, 256, 0, stream>>>(hsA, Woutf, b_out, out);
}
